// CU_16896401342567
// MI455X (gfx1250) — matrix operand order repaired and checked by reading, not compiled, not run on hardware
//
#include <hip/hip_runtime.h>
#include <hip/hip_bf16.h>

typedef __attribute__((ext_vector_type(16))) _Float16 v16h;
typedef __attribute__((ext_vector_type(8)))  float    v8f;

#define HH 256
#define WW 256
#define PLANE (HH*WW)   // 65536

// ---------------------------------------------------------------------------
// Kernel 1: fused depthwise 7x7 + 5x5 + 3x3 (as one effective 7x7 stencil)
//           + bias sum, writes xm into `xm` (aliased to d_out), and emits
//           deterministic per-block partial sums for the SE average pool.
// Grid: 65536 blocks (16 n * 64 c * 64 tiles), 256 threads.
// ---------------------------------------------------------------------------
__global__ void dwconv_kernel(const float* __restrict__ x,
                              const float* __restrict__ w7, const float* __restrict__ b7,
                              const float* __restrict__ w5, const float* __restrict__ b5,
                              const float* __restrict__ w3, const float* __restrict__ b3,
                              float* __restrict__ xm,
                              float* __restrict__ partial)
{
    __shared__ float sPatch[38 * 40];   // 38x38 patch, row stride 40 (bank spread)
    __shared__ float sWeff[49];
    __shared__ float sBias;
    __shared__ float sRed[256];

    const int b    = blockIdx.x;
    const int n    = b >> 12;          // / 4096
    const int rem  = b & 4095;
    const int c    = rem >> 6;
    const int tile = rem & 63;
    const int hbase = (tile >> 3) << 5;
    const int wbase = (tile & 7)  << 5;
    const int tid = threadIdx.x;

    // Effective 7x7 weights = w7 + zero-padded w5 + zero-padded w3
    if (tid < 49) {
        int i = tid / 7, j = tid % 7;
        float v = w7[c * 49 + tid];
        if (i >= 1 && i <= 5 && j >= 1 && j <= 5) v += w5[c * 25 + (i - 1) * 5 + (j - 1)];
        if (i >= 2 && i <= 4 && j >= 2 && j <= 4) v += w3[c * 9  + (i - 2) * 3 + (j - 2)];
        sWeff[tid] = v;
    } else if (tid == 49) {
        sBias = b7[c] + b5[c] + b3[c];
    }

    // Stage 38x38 input patch (pad=3, zero OOB)
    const float* xp = x + (size_t)(n * 64 + c) * PLANE;
    for (int idx = tid; idx < 38 * 38; idx += 256) {
        int pr = idx / 38, pc = idx % 38;
        int h = hbase - 3 + pr, w = wbase - 3 + pc;
        float v = 0.0f;
        if ((unsigned)h < (unsigned)HH && (unsigned)w < (unsigned)WW)
            v = xp[h * WW + w];
        sPatch[pr * 40 + pc] = v;
    }
    __syncthreads();

    float* outp = xm + (size_t)(n * 64 + c) * PLANE;
    const float bsum = sBias;
    float lsum = 0.0f;
    #pragma unroll
    for (int k = 0; k < 4; ++k) {
        int pix = tid + k * 256;           // 0..1023 over 32x32 tile
        int r  = pix >> 5;
        int cc = pix & 31;
        float acc = bsum;
        #pragma unroll
        for (int i = 0; i < 7; ++i)
            #pragma unroll
            for (int j = 0; j < 7; ++j)
                acc = fmaf(sWeff[i * 7 + j], sPatch[(r + i) * 40 + cc + j], acc);
        outp[(hbase + r) * WW + (wbase + cc)] = acc;
        lsum += acc;
    }

    // Deterministic block reduction for avg-pool partials
    sRed[tid] = lsum;
    __syncthreads();
    for (int s = 128; s > 0; s >>= 1) {
        if (tid < s) sRed[tid] += sRed[tid + s];
        __syncthreads();
    }
    if (tid == 0) partial[(n * 64 + c) * 64 + tile] = sRed[0];
}

// ---------------------------------------------------------------------------
// Kernel 2: SE head. Reduce partials -> mean -> fc1/relu -> fc2/sigmoid.
// Grid: 16 blocks (one per n), 64 threads (one per channel).
// ---------------------------------------------------------------------------
__global__ void se_kernel(const float* __restrict__ partial,
                          const float* __restrict__ fc1_w, const float* __restrict__ fc1_b,
                          const float* __restrict__ fc2_w, const float* __restrict__ fc2_b,
                          float* __restrict__ scale)
{
    __shared__ float sMean[64];
    __shared__ float sHid[4];
    const int n = blockIdx.x;
    const int c = threadIdx.x;

    float s = 0.0f;
    const float* p = partial + (n * 64 + c) * 64;
    #pragma unroll 8
    for (int t = 0; t < 64; ++t) s += p[t];
    sMean[c] = s * (1.0f / (float)PLANE);
    __syncthreads();

    if (c < 4) {
        float h = fc1_b[c];
        #pragma unroll 8
        for (int k = 0; k < 64; ++k) h = fmaf(sMean[k], fc1_w[c * 64 + k], h);
        sHid[c] = fmaxf(h, 0.0f);
    }
    __syncthreads();

    float z = fc2_b[c];
    #pragma unroll
    for (int j = 0; j < 4; ++j) z = fmaf(sHid[j], fc2_w[c * 4 + j], z);
    scale[n * 64 + c] = 1.0f / (1.0f + __expf(-z));
}

// ---------------------------------------------------------------------------
// Kernel 3: 1x1 conv as WMMA GEMM + bias + residual, in-place on d_out.
//   out[n,o,p] = sum_c (W[o,c]*s[n,c]) * xm[n,c,p] + b[o] + x[n,o,p]
// Per block: build per-sample f16 weight matrix Wn (64x64) in LDS.
// Per wave: one 16-pixel column; A = Wn tile (16o x 32c), B = xm tile
// (32c x 16p); K=64 via two chained v_wmma_f32_16x16x32_f16; 4 o-tiles.
// In-place is safe: a wave reads all 64 channels of its pixel column
// (both K fragments feed every o-tile's accumulator) before any store.
// Grid: 8192 blocks (16 n * 512 groups of 128 pixels), 256 threads.
// ---------------------------------------------------------------------------
__global__ void pw_wmma_kernel(const float* __restrict__ x,
                               const float* __restrict__ w1x1,
                               const float* __restrict__ b1x1,
                               const float* __restrict__ scale,
                               float* __restrict__ io /* xm in, out */)
{
    __shared__ _Float16 sW[64 * 66];  // [o][c], row stride 66 halfs (132B)
    __shared__ float    sB[64];

    const int n    = blockIdx.x >> 9;     // / 512
    const int grp  = blockIdx.x & 511;
    const int tid  = threadIdx.x;
    const int wave = tid >> 5;
    const int lane = tid & 31;

    // Fold SE scale into the weight matrix: Wn[o][c] = W[o][c] * s[n][c]
    const float* sc = scale + n * 64;
    for (int idx = tid; idx < 4096; idx += 256) {
        int o = idx >> 6, c = idx & 63;
        sW[o * 66 + c] = (_Float16)(w1x1[o * 64 + c] * sc[c]);
    }
    if (tid < 64) sB[tid] = b1x1[tid];
    __syncthreads();

    const size_t nbase   = (size_t)n * 64 * PLANE;
    const int    pixbase = grp * 128 + wave * 16;
    const int    p       = lane & 15;
    const int    half    = lane >> 4;
    const size_t pixcol  = nbase + (size_t)(pixbase + p);

    // B fragments (activations): same K order as A:
    //   half 0: K=0..7 (v0-3), K=16..23 (v4-7); half 1: K=8..15 (v0-3), K=24..31 (v4-7)
    v16h bfrag[2];
    #pragma unroll
    for (int kt = 0; kt < 2; ++kt) {
        #pragma unroll
        for (int v = 0; v < 8; ++v) {
            int k = kt * 32 + ((v < 4) ? (2 * v) : (16 + 2 * (v - 4))) + 8 * half;
            bfrag[kt][2 * v]     = (_Float16)io[pixcol + (size_t)k * PLANE];
            bfrag[kt][2 * v + 1] = (_Float16)io[pixcol + (size_t)(k + 1) * PLANE];
        }
    }

    #pragma unroll
    for (int ot = 0; ot < 4; ++ot) {
        v8f acc = {};
        #pragma unroll
        for (int kt = 0; kt < 2; ++kt) {
            // A fragment (weights): ISA 16-bit A 16x32 layout:
            //   lane m = lane%16; half 0: K=0..7 (v0-3), K=16..23 (v4-7)
            //                     half 1: K=8..15 (v0-3), K=24..31 (v4-7)
            v16h a;
            #pragma unroll
            for (int v = 0; v < 8; ++v) {
                int kk = (v < 4) ? (2 * v) : (16 + 2 * (v - 4));
                int k  = kt * 32 + kk + 8 * half;
                a[2 * v]     = sW[(ot * 16 + p) * 66 + k];
                a[2 * v + 1] = sW[(ot * 16 + p) * 66 + k + 1];
            }
            acc = __builtin_amdgcn_wmma_f32_16x16x32_f16(
                /*neg_a=*/false, a, /*neg_b=*/false, bfrag[kt],
                /*c_mod=*/(short)0, acc, /*reuse_a=*/false, /*reuse_b=*/false);
        }
        // D layout: lane holds pixel p = lane%16; VGPR r -> o = r + 8*half
        #pragma unroll
        for (int r = 0; r < 8; ++r) {
            int o = ot * 16 + r + 8 * half;
            size_t idx = nbase + (size_t)o * PLANE + (size_t)(pixbase + p);
            io[idx] = acc[r] + sB[o] + x[idx];
        }
    }
}

// ---------------------------------------------------------------------------
extern "C" void kernel_launch(void* const* d_in, const int* in_sizes, int n_in,
                              void* d_out, int out_size, void* d_ws, size_t ws_size,
                              hipStream_t stream) {
    const float* x    = (const float*)d_in[0];
    const float* w7   = (const float*)d_in[1];
    const float* b7   = (const float*)d_in[2];
    const float* w5   = (const float*)d_in[3];
    const float* b5   = (const float*)d_in[4];
    const float* w3   = (const float*)d_in[5];
    const float* b3   = (const float*)d_in[6];
    const float* fc1w = (const float*)d_in[7];
    const float* fc1b = (const float*)d_in[8];
    const float* fc2w = (const float*)d_in[9];
    const float* fc2b = (const float*)d_in[10];
    const float* w1   = (const float*)d_in[11];
    const float* b1   = (const float*)d_in[12];

    float* out      = (float*)d_out;
    float* partial  = (float*)d_ws;              // 16*64*64 floats = 256 KB
    float* scalebuf = partial + 16 * 64 * 64;    // 16*64 floats

    // Pass 1: depthwise stencils -> xm (staged in d_out) + pool partials
    dwconv_kernel<<<dim3(16 * 64 * 64), dim3(256), 0, stream>>>(
        x, w7, b7, w5, b5, w3, b3, out, partial);

    // Pass 2: SE head -> per-(n,c) scales
    se_kernel<<<dim3(16), dim3(64), 0, stream>>>(
        partial, fc1w, fc1b, fc2w, fc2b, scalebuf);

    // Pass 3: WMMA 1x1 conv + bias + residual, in-place on d_out
    pw_wmma_kernel<<<dim3(16 * 512), dim3(256), 0, stream>>>(
        x, w1, b1, scalebuf, out);
}
